// MoE_57475252355401
// MI455X (gfx1250) — compile-verified, matrix operand order checked by reading, not run on hardware
//
#include <hip/hip_runtime.h>
#include <math.h>

// Problem constants (match reference)
constexpr int kB   = 4;
constexpr int kN   = 2048;
constexpr int kD   = 1024;
constexpr int kE   = 16;
constexpr int kDFF = 4096;
constexpr int kCAP = 256;
constexpr int kSINK = 6;

// POD vector types (HIP's uint4/float4 are classes -> not union-safe)
typedef __attribute__((ext_vector_type(4)))  unsigned int u32x4;
typedef __attribute__((ext_vector_type(4)))  float        f32x4;
typedef __attribute__((ext_vector_type(8)))  float        v8f;
typedef __attribute__((ext_vector_type(16))) __bf16       v16bf;

union Frag16 { u32x4 u[2]; v16bf v; };

__device__ __forceinline__ unsigned short f2bf(float f) {
  unsigned int u = __float_as_uint(f);
  u = (u + 0x7FFFu + ((u >> 16) & 1u)) >> 16;   // round-to-nearest-even
  return (unsigned short)u;
}

// packed fp32x2 -> bf16x2 (one VALU op; RNE)
__device__ __forceinline__ unsigned cvt_pk_bf16(float a, float b) {
  unsigned r;
  asm volatile("v_cvt_pk_bf16_f32 %0, %1, %2" : "=v"(r) : "v"(a), "v"(b));
  return r;
}

__device__ __forceinline__ float gelu_tanh(float x) {
  const float c = 0.7978845608028654f;          // sqrt(2/pi)
  float x3 = x * x * x;
  return 0.5f * x * (1.0f + tanhf(c * (x + 0.044715f * x3)));
}

// ---------------------------------------------------------------------------
// 0) zero the output
__global__ void zero_kernel(float* __restrict__ p, long n) {
  long i = (long)blockIdx.x * blockDim.x + threadIdx.x;
  if (i < n) p[i] = 0.0f;
}

// ---------------------------------------------------------------------------
// 1) gating logits: logits[b][e][n] = dot(x[b,n,:], gq[e,:]) * exp(-log_temp)
__global__ __launch_bounds__(256) void gate_kernel(
    const float* __restrict__ x, const float* __restrict__ gq,
    const float* __restrict__ log_temp, float* __restrict__ logits)
{
  __shared__ float gqs[kE][256];
  int tid = threadIdx.x;
  int b   = blockIdx.x / (kN / 256);
  int n   = (blockIdx.x % (kN / 256)) * 256 + tid;

  float acc[kE];
  #pragma unroll
  for (int e = 0; e < kE; ++e) acc[e] = 0.0f;

  const float* xr = x + ((size_t)b * kN + n) * kD;
  for (int dt = 0; dt < kD; dt += 256) {
    __syncthreads();
    #pragma unroll
    for (int e = 0; e < kE; ++e) gqs[e][tid] = gq[e * kD + dt + tid];
    __syncthreads();
    for (int d = 0; d < 256; ++d) {
      float xv = xr[dt + d];
      #pragma unroll
      for (int e = 0; e < kE; ++e) acc[e] = fmaf(xv, gqs[e][d], acc[e]);
    }
  }
  float inv_temp = expf(-log_temp[0]);
  #pragma unroll
  for (int e = 0; e < kE; ++e)
    logits[((size_t)b * kE + e) * kN + n] = acc[e] * inv_temp;
}

// ---------------------------------------------------------------------------
// 2a) subtract logsumexp over experts (axis e, size 16), per (b,n)
__global__ void norm_e_kernel(float* __restrict__ t) {
  int tid = blockIdx.x * blockDim.x + threadIdx.x;   // over B*N
  int b = tid / kN, n = tid % kN;
  float* base = t + (size_t)b * kE * kN + n;
  float v[kE];
  float m = -1e30f;
  #pragma unroll
  for (int e = 0; e < kE; ++e) { v[e] = base[(size_t)e * kN]; m = fmaxf(m, v[e]); }
  float s = 0.0f;
  #pragma unroll
  for (int e = 0; e < kE; ++e) s += expf(v[e] - m);
  float lse = m + logf(s);
  #pragma unroll
  for (int e = 0; e < kE; ++e) base[(size_t)e * kN] = v[e] - lse;
}

// 2b) subtract logsumexp over tokens (axis n, size 2048), block per (b,e)
__global__ __launch_bounds__(256) void norm_n_kernel(float* __restrict__ t) {
  __shared__ float red[256];
  float* row = t + (size_t)blockIdx.x * kN;
  int tid = threadIdx.x;
  float v[8];
  float m = -1e30f;
  #pragma unroll
  for (int i = 0; i < 8; ++i) { v[i] = row[tid + i * 256]; m = fmaxf(m, v[i]); }
  red[tid] = m; __syncthreads();
  for (int s = 128; s > 0; s >>= 1) {
    if (tid < s) red[tid] = fmaxf(red[tid], red[tid + s]);
    __syncthreads();
  }
  m = red[0]; __syncthreads();
  float sum = 0.0f;
  #pragma unroll
  for (int i = 0; i < 8; ++i) sum += expf(v[i] - m);
  red[tid] = sum; __syncthreads();
  for (int s = 128; s > 0; s >>= 1) {
    if (tid < s) red[tid] += red[tid + s];
    __syncthreads();
  }
  float lse = m + logf(red[0]);
  #pragma unroll
  for (int i = 0; i < 8; ++i) row[tid + i * 256] = v[i] - lse;
}

// ---------------------------------------------------------------------------
// 3) expert-choice top-k: block per (b,e); gates = exp(selected logit)
__global__ __launch_bounds__(256) void topk_kernel(
    const float* __restrict__ t, float* __restrict__ gates, int* __restrict__ idxb)
{
  __shared__ float vals[kN];
  __shared__ float rv[256];
  __shared__ int   ri[256];
  int g = blockIdx.x, tid = threadIdx.x;
  const float* row = t + (size_t)g * kN;
  #pragma unroll
  for (int i = 0; i < 8; ++i) vals[tid + i * 256] = row[tid + i * 256];
  __syncthreads();
  for (int c = 0; c < kCAP; ++c) {
    float bm = -1e30f; int bi = tid * 8;
    #pragma unroll
    for (int i = 0; i < 8; ++i) {
      int j = tid * 8 + i; float v = vals[j];
      if (v > bm) { bm = v; bi = j; }
    }
    rv[tid] = bm; ri[tid] = bi; __syncthreads();
    for (int s = 128; s > 0; s >>= 1) {
      if (tid < s) {
        if (rv[tid + s] > rv[tid] ||
            (rv[tid + s] == rv[tid] && ri[tid + s] < ri[tid])) {
          rv[tid] = rv[tid + s]; ri[tid] = ri[tid + s];
        }
      }
      __syncthreads();
    }
    if (tid == 0) {
      gates[(size_t)g * kCAP + c] = expf(rv[0]);
      idxb[(size_t)g * kCAP + c]  = ri[0];
      vals[ri[0]] = -1e30f;
    }
    __syncthreads();
  }
}

// ---------------------------------------------------------------------------
// WMMA GEMM tiles: 128x128 per block (8 waves of wave32), K-step 32 (bf16),
// ping-pong double-buffered LDS.
constexpr int TM = 128, TN = 128, TK = 32;
constexpr int LDA_ = 40;   // ushort stride, 80 B rows (16 B aligned, bank-spread)
constexpr int LDB_ = 40;

// 4) GEMM1: h[g, 0:256, 0:4096] = gelu(gather(x, idx) @ w1[e] + b1[e]) (bf16 out)
//    group order: e = g/4 (same-expert blocks adjacent in dispatch -> L2 reuse)
__global__ __launch_bounds__(256) void gemm1_kernel(
    const float* __restrict__ x, const float* __restrict__ w1,
    const float* __restrict__ b1, const int* __restrict__ idxb,
    unsigned short* __restrict__ h)
{
  __shared__ __align__(16) unsigned short As[2][TM][LDA_];
  __shared__ __align__(16) unsigned short Bs[2][TN][LDB_];
  __shared__ int toks[TM];

  int g = blockIdx.z, e = g >> 2, b = g & 3;
  int gidx = b * kE + e;                       // index into idx/gates tables
  int n0 = blockIdx.x * TN, m0 = blockIdx.y * TM;
  int tid = threadIdx.x;
  if (tid < TM) toks[tid] = idxb[(size_t)gidx * kCAP + m0 + tid];
  __syncthreads();

  int lane  = tid & 31, wave = tid >> 5;
  int wm = wave >> 2, wn = wave & 3;          // 2x4 wave grid: 64x32 per wave
  int lhalf = lane >> 4, lmod = lane & 15;
  int ar = tid >> 3, aq = tid & 7;            // A loads: 32 rows x 8 float4
  int bk = tid >> 5, bn = (tid & 31) * 4;     // B loads: 8 k x 32 float4

  v8f acc[4][2];
  #pragma unroll
  for (int mi = 0; mi < 4; ++mi)
    #pragma unroll
    for (int ni = 0; ni < 2; ++ni) acc[mi][ni] = {};

  // prologue: stage tile k0=0 in registers
  f32x4 areg[4], breg[4];
  #pragma unroll
  for (int p = 0; p < 4; ++p) {
    int row = p * 32 + ar;
    areg[p] = *(const f32x4*)(x + ((size_t)b * kN + toks[row]) * kD + aq * 4);
    int k = p * 8 + bk;
    breg[p] = *(const f32x4*)(w1 + ((size_t)e * kD + k) * kDFF + n0 + bn);
  }

  int buf = 0;
  for (int k0 = 0; k0 < kD; k0 += TK) {
    // commit staged tile to LDS[buf]
    #pragma unroll
    for (int p = 0; p < 4; ++p) {
      int row = p * 32 + ar;
      *(unsigned*)&As[buf][row][aq * 4 + 0] = cvt_pk_bf16(areg[p].x, areg[p].y);
      *(unsigned*)&As[buf][row][aq * 4 + 2] = cvt_pk_bf16(areg[p].z, areg[p].w);
      int k = p * 8 + bk;
      Bs[buf][bn + 0][k] = f2bf(breg[p].x);
      Bs[buf][bn + 1][k] = f2bf(breg[p].y);
      Bs[buf][bn + 2][k] = f2bf(breg[p].z);
      Bs[buf][bn + 3][k] = f2bf(breg[p].w);
    }
    // prefetch tile k0+TK into registers (overlaps with compute below)
    int k1 = k0 + TK;
    if (k1 < kD) {
      #pragma unroll
      for (int p = 0; p < 4; ++p) {
        int row = p * 32 + ar;
        areg[p] = *(const f32x4*)(x + ((size_t)b * kN + toks[row]) * kD + k1 + aq * 4);
        int k = p * 8 + bk;
        breg[p] = *(const f32x4*)(w1 + ((size_t)e * kD + (k1 + k)) * kDFF + n0 + bn);
      }
    }
    __syncthreads();

    Frag16 a[4], bf[2];
    #pragma unroll
    for (int mi = 0; mi < 4; ++mi) {
      int row = wm * 64 + mi * 16 + lmod;
      a[mi].u[0] = *(const u32x4*)&As[buf][row][lhalf * 8];
      a[mi].u[1] = *(const u32x4*)&As[buf][row][16 + lhalf * 8];
    }
    #pragma unroll
    for (int ni = 0; ni < 2; ++ni) {
      int col = wn * 32 + ni * 16 + lmod;
      bf[ni].u[0] = *(const u32x4*)&Bs[buf][col][lhalf * 8];
      bf[ni].u[1] = *(const u32x4*)&Bs[buf][col][16 + lhalf * 8];
    }
    #pragma unroll
    for (int mi = 0; mi < 4; ++mi)
      #pragma unroll
      for (int ni = 0; ni < 2; ++ni)
        acc[mi][ni] = __builtin_amdgcn_wmma_f32_16x16x32_bf16(
            false, a[mi].v, false, bf[ni].v, (short)0, acc[mi][ni], false, false);
    buf ^= 1;
  }

  #pragma unroll
  for (int mi = 0; mi < 4; ++mi) {
    int m = m0 + wm * 64 + mi * 16 + lhalf * 8;
    #pragma unroll
    for (int ni = 0; ni < 2; ++ni) {
      int n = n0 + wn * 32 + ni * 16 + lmod;
      float bias = b1[e * kDFF + n];
      #pragma unroll
      for (int r = 0; r < 8; ++r) {
        float val = gelu_tanh(acc[mi][ni][r] + bias);
        h[((size_t)g * kCAP + (m + r)) * kDFF + n] = f2bf(val);
      }
    }
  }
}

// 5) GEMM2: out[b, idx, :] += gate * (h @ w2[e] + b2[e])   (atomic scatter)
//    A-tile (bf16 h) streams via GLOBAL_LOAD_ASYNC_TO_LDS_B128 (ASYNCcnt).
//    Async issue happens AFTER the barrier so no wave can still be reading
//    the destination buffer (reads of buf^1 finished before barrier i).
__global__ __launch_bounds__(256) void gemm2_kernel(
    const unsigned short* __restrict__ h, const float* __restrict__ w2,
    const float* __restrict__ b2, const int* __restrict__ idxb,
    const float* __restrict__ gates, float* __restrict__ out)
{
  __shared__ __align__(16) unsigned short As[2][TM][LDA_];
  __shared__ __align__(16) unsigned short Bs[2][TN][LDB_];
  __shared__ int   toks[TM];
  __shared__ float gvals[TM];

  int g = blockIdx.z, e = g >> 2, b = g & 3;
  int gidx = b * kE + e;
  int n0 = blockIdx.x * TN, m0 = blockIdx.y * TM;
  int tid = threadIdx.x;
  if (tid < TM) {
    toks[tid]  = idxb[(size_t)gidx * kCAP + m0 + tid];
    gvals[tid] = gates[(size_t)gidx * kCAP + m0 + tid];
  }
  __syncthreads();

  int lane  = tid & 31, wave = tid >> 5;
  int wm = wave >> 2, wn = wave & 3;
  int lhalf = lane >> 4, lmod = lane & 15;
  int bk = tid >> 5, bn = (tid & 31) * 4;

  // async A-tile copy: 2 x 16B per thread per K-step, straight global->LDS
  auto issue_asyncA = [&](int k0, int bsel) {
    #pragma unroll
    for (int p = 0; p < 2; ++p) {
      int index = p * 256 + tid;
      int row = index >> 2, q = index & 3;
      const unsigned short* gp =
          h + ((size_t)g * kCAP + m0 + row) * kDFF + k0 + q * 8;
      unsigned lds = (unsigned)(size_t)&As[bsel][row][q * 8];
      asm volatile("global_load_async_to_lds_b128 %0, %1, off"
                   :: "v"(lds), "v"(gp) : "memory");
    }
  };

  v8f acc[4][2];
  #pragma unroll
  for (int mi = 0; mi < 4; ++mi)
    #pragma unroll
    for (int ni = 0; ni < 2; ++ni) acc[mi][ni] = {};

  // prologue: async A tile 0 -> As[0]; stage B tile 0 in registers
  issue_asyncA(0, 0);
  f32x4 breg[4];
  #pragma unroll
  for (int p = 0; p < 4; ++p) {
    int k = p * 8 + bk;
    breg[p] = *(const f32x4*)(w2 + ((size_t)e * kDFF + k) * kD + n0 + bn);
  }

  int buf = 0;
  for (int k0 = 0; k0 < kDFF; k0 += TK) {
    #pragma unroll
    for (int p = 0; p < 4; ++p) {            // commit staged B tile
      int k = p * 8 + bk;
      Bs[buf][bn + 0][k] = f2bf(breg[p].x);
      Bs[buf][bn + 1][k] = f2bf(breg[p].y);
      Bs[buf][bn + 2][k] = f2bf(breg[p].z);
      Bs[buf][bn + 3][k] = f2bf(breg[p].w);
    }
    int k1 = k0 + TK;
    if (k1 < kDFF) {                          // prefetch next B into registers
      #pragma unroll
      for (int p = 0; p < 4; ++p) {
        int k = p * 8 + bk;
        breg[p] = *(const f32x4*)(w2 + ((size_t)e * kDFF + (k1 + k)) * kD + n0 + bn);
      }
    }
    // only this tile's 2 asyncs are outstanding here -> cheap full wait
    asm volatile("s_wait_asynccnt 0x0" ::: "memory");
    __syncthreads();
    if (k1 < kDFF)                            // safe: all reads of buf^1 done
      issue_asyncA(k1, buf ^ 1);              // overlaps with WMMA below

    Frag16 a[4], bf[2];
    #pragma unroll
    for (int mi = 0; mi < 4; ++mi) {
      int row = wm * 64 + mi * 16 + lmod;
      a[mi].u[0] = *(const u32x4*)&As[buf][row][lhalf * 8];
      a[mi].u[1] = *(const u32x4*)&As[buf][row][16 + lhalf * 8];
    }
    #pragma unroll
    for (int ni = 0; ni < 2; ++ni) {
      int col = wn * 32 + ni * 16 + lmod;
      bf[ni].u[0] = *(const u32x4*)&Bs[buf][col][lhalf * 8];
      bf[ni].u[1] = *(const u32x4*)&Bs[buf][col][16 + lhalf * 8];
    }
    #pragma unroll
    for (int mi = 0; mi < 4; ++mi)
      #pragma unroll
      for (int ni = 0; ni < 2; ++ni)
        acc[mi][ni] = __builtin_amdgcn_wmma_f32_16x16x32_bf16(
            false, a[mi].v, false, bf[ni].v, (short)0, acc[mi][ni], false, false);
    buf ^= 1;
  }

  #pragma unroll
  for (int mi = 0; mi < 4; ++mi) {
    int lr = wm * 64 + mi * 16 + lhalf * 8;   // local row (0..127), +r below
    #pragma unroll
    for (int ni = 0; ni < 2; ++ni) {
      int n = n0 + wn * 32 + ni * 16 + lmod;
      float bias = b2[e * kD + n];
      #pragma unroll
      for (int r = 0; r < 8; ++r) {
        int tok  = toks[lr + r];
        float gv = gvals[lr + r];
        float val = gv * (acc[mi][ni][r] + bias);
        unsafeAtomicAdd(&out[((size_t)b * kN + tok) * kD + n], val);
      }
    }
  }
}

// ---------------------------------------------------------------------------
extern "C" void kernel_launch(void* const* d_in, const int* in_sizes, int n_in,
                              void* d_out, int out_size, void* d_ws, size_t ws_size,
                              hipStream_t stream) {
  const float* x        = (const float*)d_in[0];
  const float* gq       = (const float*)d_in[1];
  const float* log_temp = (const float*)d_in[2];
  const float* w1       = (const float*)d_in[3];
  const float* b1       = (const float*)d_in[4];
  const float* w2       = (const float*)d_in[5];
  const float* b2       = (const float*)d_in[6];
  float* out = (float*)d_out;

  // workspace layout
  char* ws = (char*)d_ws;
  float* logits = (float*)ws;                                  ws += (size_t)kB * kE * kN * 4;
  float* gates  = (float*)ws;                                  ws += (size_t)kB * kE * kCAP * 4;
  int*   idxb   = (int*)ws;                                    ws += (size_t)kB * kE * kCAP * 4;
  unsigned short* h = (unsigned short*)ws;  // bf16 [64][256][4096], 128 MB

  long outN = (long)kB * kN * kD;
  zero_kernel<<<(outN + 255) / 256, 256, 0, stream>>>(out, outN);

  gate_kernel<<<kB * kN / 256, 256, 0, stream>>>(x, gq, log_temp, logits);

  for (int it = 0; it < kSINK; ++it) {
    norm_e_kernel<<<kB * kN / 256, 256, 0, stream>>>(logits);
    norm_n_kernel<<<kB * kE, 256, 0, stream>>>(logits);
  }

  topk_kernel<<<kB * kE, 256, 0, stream>>>(logits, gates, idxb);

  gemm1_kernel<<<dim3(kDFF / TN, kCAP / TM, kB * kE), 256, 0, stream>>>(
      x, w1, b1, idxb, h);

  gemm2_kernel<<<dim3(kD / TN, kCAP / TM, kB * kE), 256, 0, stream>>>(
      h, w2, b2, idxb, gates, out);
}
